// BlockDiffusionDiT_9637906612382
// MI455X (gfx1250) — compile-verified, matrix operand order checked by reading, not run on hardware
//
#include <hip/hip_runtime.h>
#include <hip/hip_bf16.h>
#include <math.h>

// ---------------------------------------------------------------------------
// BlockDiffusionDiT block for MI455X (gfx1250)
//   - bf16 end-to-end activations, f32 WMMA accumulation
//   - 128x128 GEMM tiles, 8 v_wmma_f32_16x16x32_bf16 per wave per K-step
//   - double-buffered GLOBAL_LOAD_ASYNC_TO_LDS_B128 staging (in-order
//     ASYNCcnt retirement -> s_wait_asynccnt 0x4 hides copy latency)
//   - WMMA flash attention with analytic block-diff mask (tile-level skip)
// ---------------------------------------------------------------------------

typedef __attribute__((ext_vector_type(16))) __bf16 v16bf;
typedef __attribute__((ext_vector_type(8)))  float  v8f;

#define D_MODEL 1024
#define NHEAD   16
#define HEADD   64
#define SEQ     2048   // 2*N
#define NB      2
#define LCOND   77
#define MODW    (9 * D_MODEL)

// ------------------------------ helpers ------------------------------------

// CDNA5 16-bit A/B fragment: lane L holds row/col (L&15); K elements:
//   kbase = (L>>4)*8 ; elems 0..7 -> K=kbase..+7 ; elems 8..15 -> K=kbase+16..+23
// Both halves are contiguous 16B runs -> two b128 loads (LDS or global).
__device__ inline v16bf ld_frag(const __bf16* p, int kb) {
  union { v16bf v; uint4 u[2]; } r;
  r.u[0] = *(const uint4*)(p + kb);
  r.u[1] = *(const uint4*)(p + kb + 16);
  return r.v;
}

__device__ inline v8f wmma_bf16(v16bf a, v16bf b, v8f c) {
  return __builtin_amdgcn_wmma_f32_16x16x32_bf16(false, a, false, b,
                                                 (short)0, c, false, false);
}

// one async wave-copy: each lane moves 16B global -> LDS (ASYNCcnt-tracked)
__device__ inline void async_copy16(const __bf16* src, __bf16* lds_dst) {
  unsigned d = (unsigned)(size_t)lds_dst;   // low 32 bits = LDS byte address
  asm volatile("global_load_async_to_lds_b128 %0, %1, off"
               :: "v"(d), "v"(src) : "memory");
}
__device__ inline void async_wait0() {
  asm volatile("s_wait_asynccnt 0x0" ::: "memory");
}
__device__ inline void async_wait4() {   // allow 4 in flight (next tile)
  asm volatile("s_wait_asynccnt 0x4" ::: "memory");
}

__device__ inline float redmax16(float v) {
#pragma unroll
  for (int m = 8; m >= 1; m >>= 1) v = fmaxf(v, __shfl_xor(v, m, 16));
  return v;
}
__device__ inline float redsum16(float v) {
#pragma unroll
  for (int m = 8; m >= 1; m >>= 1) v += __shfl_xor(v, m, 16);
  return v;
}

// ---------------------- weight transpose + f32->bf16 -----------------------
// W[K][N] f32 (row major)  ->  Wt[N][K] bf16

__global__ __launch_bounds__(256) void transpose_cvt_kernel(
    const float* __restrict__ W, __bf16* __restrict__ Wt, int K, int N) {
  __shared__ float t[32][33];
  int kt = blockIdx.x * 32, nt = blockIdx.y * 32;
  int tx = threadIdx.x & 31, ty = threadIdx.x >> 5;   // ty: 0..7
#pragma unroll
  for (int i = ty; i < 32; i += 8) t[i][tx] = W[(size_t)(kt + i) * N + nt + tx];
  __syncthreads();
#pragma unroll
  for (int i = ty; i < 32; i += 8)
    Wt[(size_t)(nt + i) * K + kt + tx] = (__bf16)t[tx][i];
}

// ---------------------------- f32 -> bf16 copy -----------------------------

__global__ void cvt_bf16_kernel(const float* __restrict__ in,
                                __bf16* __restrict__ out, int n) {
  int i = blockIdx.x * blockDim.x + threadIdx.x;
  if (i < n) out[i] = (__bf16)in[i];
}

// -------------------------- adaLN modulation -------------------------------

__global__ void adaln_kernel(const float* __restrict__ cg,
                             const float* __restrict__ W,
                             const float* __restrict__ bias,
                             float* __restrict__ mod) {
  int idx = blockIdx.x * blockDim.x + threadIdx.x;
  if (idx >= NB * MODW) return;
  int b = idx / MODW, j = idx % MODW;
  float acc = bias[j];
  const float* c = cg + b * D_MODEL;
  for (int k = 0; k < D_MODEL; ++k) acc += c[k] * W[(size_t)k * MODW + j];
  mod[idx] = acc;
}

// ------------------- LayerNorm + adaLN scale/shift -> bf16 ------------------

__global__ __launch_bounds__(256) void ln_mod_kernel(
    const float* __restrict__ x, const float* __restrict__ w,
    const float* __restrict__ mod, int off_sh, int off_sc,
    __bf16* __restrict__ out) {
  int row = blockIdx.x;
  int tid = threadIdx.x;
  int bidx = row / SEQ;
  const float* xr = x + (size_t)row * D_MODEL;
  __shared__ float s1[256], s2[256];
  float a = 0.f, b = 0.f;
  for (int c = tid; c < D_MODEL; c += 256) { float v = xr[c]; a += v; b += v * v; }
  s1[tid] = a; s2[tid] = b;
  __syncthreads();
  for (int off = 128; off > 0; off >>= 1) {
    if (tid < off) { s1[tid] += s1[tid + off]; s2[tid] += s2[tid + off]; }
    __syncthreads();
  }
  float mu   = s1[0] * (1.f / D_MODEL);
  float var  = s2[0] * (1.f / D_MODEL) - mu * mu;
  float rstd = rsqrtf(var + 1e-5f);
  const float* mrow = mod + bidx * MODW;
  __bf16* orow = out + (size_t)row * D_MODEL;
  for (int c = tid; c < D_MODEL; c += 256) {
    float sc = mrow[off_sc + c], sh = mrow[off_sh + c];
    orow[c] = (__bf16)((xr[c] - mu) * rstd * w[c] * (1.f + sc) + sh);
  }
}

// --------------------------------- RoPE ------------------------------------
// qkv (bf16, (B*s, 3D)) -> qr, kr (bf16, (B*s, D)); pos = s mod 1024

__global__ void rope_kernel(const __bf16* __restrict__ qkv,
                            __bf16* __restrict__ qr, __bf16* __restrict__ kr) {
  int idx = blockIdx.x * blockDim.x + threadIdx.x;
  int d = idx & 31;
  int h = (idx >> 5) & 15;
  int s = (idx >> 9) & 2047;
  int b = idx >> 20;
  if (b >= NB) return;
  int pos = s & 1023;
  float inv = __expf(-(2.f * (float)d / 64.f) * 9.210340371976184f); // ln(1e4)
  float ang = (float)pos * inv;
  float c = __cosf(ang), sn = __sinf(ang);
  size_t row  = (size_t)b * SEQ + s;
  size_t base = row * (3 * D_MODEL) + h * HEADD;
  float q1 = (float)qkv[base + d],           q2 = (float)qkv[base + 32 + d];
  float k1 = (float)qkv[base + D_MODEL + d], k2 = (float)qkv[base + D_MODEL + 32 + d];
  size_t ob = row * D_MODEL + h * HEADD;
  qr[ob + d]      = (__bf16)(q1 * c - q2 * sn);
  qr[ob + 32 + d] = (__bf16)(q2 * c + q1 * sn);
  kr[ob + d]      = (__bf16)(k1 * c - k2 * sn);
  kr[ob + 32 + d] = (__bf16)(k2 * c + k1 * sn);
}

// ------------------ bf16 WMMA GEMM (double-buffered async) ------------------
// C[M,N] = A[M,K](bf16) * Bt[N,K](bf16, pre-transposed)
// epilogue: (+bias) (gelu) (resid + gate*val); C is f32 or bf16

__global__ __launch_bounds__(256) void gemm_bf16_kernel(
    const __bf16* __restrict__ A, const __bf16* __restrict__ Bt,
    void* __restrict__ C, int M, int N, int K, int lda, int ldc,
    const float* __restrict__ bias, int do_gelu, int out_bf16,
    const float* __restrict__ resid, const float* __restrict__ gate,
    int rows_per_batch) {
  __shared__ __bf16 As[2][128][40];   // [m][k] ; 80B row stride (16B aligned)
  __shared__ __bf16 Bs[2][128][40];   // [n][k]
  int tid  = threadIdx.x;
  int lane = tid & 31;
  int wave = tid >> 5;
  int wm = (wave >> 1) * 32;       // 4 waves along M
  int wn = (wave & 1) * 64;        // 2 waves along N
  int m0 = blockIdx.x * 128;
  int n0 = blockIdx.y * 128;
  int l15 = lane & 15;
  int kb  = (lane >> 4) * 8;
  v8f zero = {};
  v8f acc[2][4];
#pragma unroll
  for (int i = 0; i < 2; ++i)
#pragma unroll
    for (int f = 0; f < 4; ++f) acc[i][f] = zero;

  // per-thread 16B staging chunks (2 for A tile, 2 for B tile)
  int r0 = tid >> 2,        e0 = (tid & 3) * 8;
  int r1 = (tid + 256) >> 2, e1 = (tid & 3) * 8;
  int ga0 = m0 + r0; ga0 = ga0 < M ? ga0 : M - 1;   // clamp (stores guarded)
  int ga1 = m0 + r1; ga1 = ga1 < M ? ga1 : M - 1;
  const __bf16* a0p = A + (size_t)ga0 * lda + e0;
  const __bf16* a1p = A + (size_t)ga1 * lda + e1;
  const __bf16* b0p = Bt + (size_t)(n0 + r0) * K + e0;
  const __bf16* b1p = Bt + (size_t)(n0 + r1) * K + e1;

  auto stage = [&](int buf, int k0) {
    async_copy16(a0p + k0, &As[buf][r0][e0]);
    async_copy16(a1p + k0, &As[buf][r1][e1]);
    async_copy16(b0p + k0, &Bs[buf][r0][e0]);
    async_copy16(b1p + k0, &Bs[buf][r1][e1]);
  };

  int nk = K / 32;
  stage(0, 0);
  for (int it = 0; it < nk; ++it) {
    int cur = it & 1;
    bool hasnext = (it + 1) < nk;          // uniform
    if (hasnext) {
      stage(1 - cur, (it + 1) * 32);       // prefetch next tile (async)
      async_wait4();                       // retire current tile only
    } else {
      async_wait0();
    }
    __syncthreads();                       // tile `it` visible to all waves
    v16bf a0 = ld_frag(&As[cur][wm + l15][0], kb);
    v16bf a1 = ld_frag(&As[cur][wm + 16 + l15][0], kb);
#pragma unroll
    for (int f = 0; f < 4; ++f) {
      v16bf bf_ = ld_frag(&Bs[cur][wn + f * 16 + l15][0], kb);
      acc[0][f] = wmma_bf16(a0, bf_, acc[0][f]);
      acc[1][f] = wmma_bf16(a1, bf_, acc[1][f]);
    }
    __syncthreads();                       // done reading buf before refill
  }
  // epilogue: C layout lane=(n&15), reg r -> m = r + 8*(lane>>4)
#pragma unroll
  for (int i = 0; i < 2; ++i) {
    int mb = m0 + wm + i * 16 + (lane >> 4) * 8;
#pragma unroll
    for (int f = 0; f < 4; ++f) {
      int n = n0 + wn + f * 16 + l15;
#pragma unroll
      for (int r = 0; r < 8; ++r) {
        int m = mb + r;
        if (m < M) {
          float v = acc[i][f][r];
          if (bias) v += bias[n];
          if (do_gelu) {
            float x3 = v * v * v;
            v = 0.5f * v * (1.f + tanhf(0.79788456f * (v + 0.044715f * x3)));
          }
          if (resid) {
            int bb = m / rows_per_batch;
            v = resid[(size_t)m * ldc + n] + gate[bb * MODW + n] * v;
          }
          if (out_bf16) ((__bf16*)C)[(size_t)m * ldc + n] = (__bf16)v;
          else          ((float*)C)[(size_t)m * ldc + n] = v;
        }
      }
    }
  }
}

// --------------------- flash attention (WMMA, 1 wave/tile) ------------------
// mode 0: block-diff self mask (tile-uniform -> skip masked 16x16 tiles)
// mode 1: cross attention, kv length skv (=77) with per-key byte mask

__global__ __launch_bounds__(32) void attn_kernel(
    const __bf16* __restrict__ Q, int ldq,
    const __bf16* __restrict__ K, int ldk,
    const __bf16* __restrict__ V, int ldv,
    __bf16* __restrict__ O, int ldo,
    int sq, int skv, int mode,
    const unsigned char* __restrict__ kvmask) {
  __shared__ __bf16 Plds[16][40];   // probabilities, A-layout staging
  __shared__ __bf16 Vt[64][40];     // V chunk transposed [dim][k]
  int lane = threadIdx.x;
  int qt = blockIdx.x, h = blockIdx.y, b = blockIdx.z;
  int col = h * HEADD;
  int l15 = lane & 15;
  int kb  = (lane >> 4) * 8;
  const __bf16* qrow = Q + (size_t)(b * sq + qt * 16 + l15) * ldq + col;
  v16bf qa0 = ld_frag(qrow, kb);
  v16bf qa1 = ld_frag(qrow + 32, kb);
  v8f o0 = {}, o1 = {}, o2 = {}, o3 = {};
  float mrun[8], lrun[8];
#pragma unroll
  for (int r = 0; r < 8; ++r) { mrun[r] = -1e30f; lrun[r] = 0.f; }
  int htiles  = sq / 32;                               // blocks per half (64)
  int nchunks = (mode == 0) ? (skv / 32) : ((skv + 31) / 32);
  for (int ck = 0; ck < nchunks; ++ck) {
    bool al[2];
    if (mode == 0) {
#pragma unroll
      for (int t = 0; t < 2; ++t) {
        int kt = ck * 2 + t;
        bool fq = qt >= htiles, fk = kt >= htiles;
        int bq = fq ? qt - htiles : qt;
        int bk = fk ? kt - htiles : kt;
        al[t] = (bq == bk && fq == fk) || (bk < bq && fk && !fq) ||
                (bk <= bq && fq && fk);
      }
    } else {
      al[0] = (ck * 32)      < skv;
      al[1] = (ck * 32 + 16) < skv;
    }
    if (!al[0] && !al[1]) continue;
    // S = (Q*K^T) * 1/sqrt(HD), two 16-key subtiles
    v8f s[2];
#pragma unroll
    for (int t = 0; t < 2; ++t) {
      if (al[t]) {
        int key  = ck * 32 + t * 16 + l15;
        int krow = key < skv ? key : skv - 1;
        const __bf16* kp = K + (size_t)(b * skv + krow) * ldk + col;
        v16bf bk0 = ld_frag(kp, kb);
        v16bf bk1 = ld_frag(kp + 32, kb);
        v8f sv = {};
        sv = wmma_bf16(qa0, bk0, sv);
        sv = wmma_bf16(qa1, bk1, sv);
#pragma unroll
        for (int r = 0; r < 8; ++r) sv[r] *= 0.125f;
        if (mode == 1) {
          bool valid = (key < skv) && (kvmask[b * skv + key] != 0);
          if (!valid) {
#pragma unroll
            for (int r = 0; r < 8; ++r) sv[r] = -1e30f;
          }
        }
        s[t] = sv;
      } else {
        v8f sv;
#pragma unroll
        for (int r = 0; r < 8; ++r) sv[r] = -1e30f;
        s[t] = sv;
      }
    }
    // online softmax (rows live in 16-lane groups)
    float alpha[8];
#pragma unroll
    for (int r = 0; r < 8; ++r) {
      float rm   = redmax16(fmaxf(s[0][r], s[1][r]));
      float mnew = fmaxf(mrun[r], rm);
      float p0 = __expf(s[0][r] - mnew);
      float p1 = __expf(s[1][r] - mnew);
      float rs = redsum16(p0 + p1);
      float a_ = __expf(mrun[r] - mnew);
      lrun[r] = lrun[r] * a_ + rs;
      mrun[r] = mnew;
      alpha[r] = a_;
      s[0][r] = p0; s[1][r] = p1;
    }
#pragma unroll
    for (int r = 0; r < 8; ++r) {
      o0[r] *= alpha[r]; o1[r] *= alpha[r];
      o2[r] *= alpha[r]; o3[r] *= alpha[r];
    }
    // stage P (C layout -> A layout) and V chunk (transposed) in LDS
    {
      int mb = (lane >> 4) * 8;
#pragma unroll
      for (int r = 0; r < 8; ++r) {
        Plds[mb + r][l15]      = (__bf16)s[0][r];
        Plds[mb + r][16 + l15] = (__bf16)s[1][r];
      }
      int key  = ck * 32 + lane;
      int vrow = key < skv ? key : skv - 1;
      const __bf16* vp = V + (size_t)(b * skv + vrow) * ldv + col;
#pragma unroll
      for (int dd = 0; dd < 64; ++dd) Vt[dd][lane] = vp[dd];
    }
    __syncthreads();
    v16bf pa = ld_frag(&Plds[l15][0], kb);
    v16bf v0 = ld_frag(&Vt[l15][0], kb);
    v16bf v1 = ld_frag(&Vt[16 + l15][0], kb);
    v16bf v2 = ld_frag(&Vt[32 + l15][0], kb);
    v16bf v3 = ld_frag(&Vt[48 + l15][0], kb);
    o0 = wmma_bf16(pa, v0, o0);
    o1 = wmma_bf16(pa, v1, o1);
    o2 = wmma_bf16(pa, v2, o2);
    o3 = wmma_bf16(pa, v3, o3);
    __syncthreads();
  }
  // normalize and store (concat-head layout (B*sq, D))
  __bf16* orow = O + (size_t)(b * sq + qt * 16) * ldo + col;
#pragma unroll
  for (int r = 0; r < 8; ++r) {
    int m = (lane >> 4) * 8 + r;
    float inv = 1.f / fmaxf(lrun[r], 1e-20f);
    orow[(size_t)m * ldo + 0  + l15] = (__bf16)(o0[r] * inv);
    orow[(size_t)m * ldo + 16 + l15] = (__bf16)(o1[r] * inv);
    orow[(size_t)m * ldo + 32 + l15] = (__bf16)(o2[r] * inv);
    orow[(size_t)m * ldo + 48 + l15] = (__bf16)(o3[r] * inv);
  }
}

// ------------------------------- launcher ----------------------------------

extern "C" void kernel_launch(void* const* d_in, const int* in_sizes, int n_in,
                              void* d_out, int out_size, void* d_ws, size_t ws_size,
                              hipStream_t stream) {
  const float* x            = (const float*)d_in[0];
  const float* cond_global  = (const float*)d_in[1];
  const float* cond_tokens  = (const float*)d_in[2];
  const float* adaLN_W      = (const float*)d_in[3];
  const float* adaLN_b      = (const float*)d_in[4];
  const float* norm1_w      = (const float*)d_in[5];
  const float* qkv_W        = (const float*)d_in[6];
  const float* attn_out_W   = (const float*)d_in[7];
  const float* normc_w      = (const float*)d_in[8];
  const float* cq_W         = (const float*)d_in[9];
  const float* ck_W         = (const float*)d_in[10];
  const float* cv_W         = (const float*)d_in[11];
  const float* co_W         = (const float*)d_in[12];
  const float* norm2_w      = (const float*)d_in[13];
  const float* mlp_W1       = (const float*)d_in[14];
  const float* mlp_b1       = (const float*)d_in[15];
  const float* mlp_W2       = (const float*)d_in[16];
  const float* mlp_b2       = (const float*)d_in[17];
  const unsigned char* cond_kv_mask = (const unsigned char*)d_in[18];
  // d_in[19] (attn_mask) is recomputed analytically on-device.

  char* w = (char*)d_ws;
  auto allocf = [&](size_t n) { float* p = (float*)w;   w += ((n * 4 + 255) / 256) * 256; return p; };
  auto allocb = [&](size_t n) { __bf16* p = (__bf16*)w; w += ((n * 2 + 255) / 256) * 256; return p; };

  float*  mod = allocf((size_t)NB * MODW);
  float*  x1  = allocf((size_t)NB * SEQ * D_MODEL);
  float*  x2  = allocf((size_t)NB * SEQ * D_MODEL);
  __bf16* lnb = allocb((size_t)NB * SEQ * D_MODEL);
  __bf16* qkv = allocb((size_t)NB * SEQ * 3 * D_MODEL);  // reused as qc later
  __bf16* qr  = allocb((size_t)NB * SEQ * D_MODEL);
  __bf16* kr  = allocb((size_t)NB * SEQ * D_MODEL);
  __bf16* ao  = allocb((size_t)NB * SEQ * D_MODEL);      // self / cross attn out
  __bf16* hb  = allocb((size_t)NB * SEQ * 4 * D_MODEL);
  __bf16* kcb = allocb((size_t)NB * LCOND * D_MODEL);
  __bf16* vcb = allocb((size_t)NB * LCOND * D_MODEL);
  __bf16* ctb = allocb((size_t)NB * LCOND * D_MODEL);
  __bf16* qkvT = allocb((size_t)3 * D_MODEL * D_MODEL);
  __bf16* aoT  = allocb((size_t)D_MODEL * D_MODEL);
  __bf16* cqT  = allocb((size_t)D_MODEL * D_MODEL);
  __bf16* ckT  = allocb((size_t)D_MODEL * D_MODEL);
  __bf16* cvT  = allocb((size_t)D_MODEL * D_MODEL);
  __bf16* coT  = allocb((size_t)D_MODEL * D_MODEL);
  __bf16* w1T  = allocb((size_t)4 * D_MODEL * D_MODEL);
  __bf16* w2T  = allocb((size_t)4 * D_MODEL * D_MODEL);
  __bf16* qc   = qkv;   // qkv fully consumed (rope + self-attn V) before qc

  int rowsAll = NB * SEQ;      // 4096
  int rowsC   = NB * LCOND;    // 154

  // 0. weight transpose+convert (f32 [K][N] -> bf16 [N][K]) + cond tokens cvt
  transpose_cvt_kernel<<<dim3(D_MODEL / 32, (3 * D_MODEL) / 32), 256, 0, stream>>>(qkv_W, qkvT, D_MODEL, 3 * D_MODEL);
  transpose_cvt_kernel<<<dim3(D_MODEL / 32, D_MODEL / 32), 256, 0, stream>>>(attn_out_W, aoT, D_MODEL, D_MODEL);
  transpose_cvt_kernel<<<dim3(D_MODEL / 32, D_MODEL / 32), 256, 0, stream>>>(cq_W, cqT, D_MODEL, D_MODEL);
  transpose_cvt_kernel<<<dim3(D_MODEL / 32, D_MODEL / 32), 256, 0, stream>>>(ck_W, ckT, D_MODEL, D_MODEL);
  transpose_cvt_kernel<<<dim3(D_MODEL / 32, D_MODEL / 32), 256, 0, stream>>>(cv_W, cvT, D_MODEL, D_MODEL);
  transpose_cvt_kernel<<<dim3(D_MODEL / 32, D_MODEL / 32), 256, 0, stream>>>(co_W, coT, D_MODEL, D_MODEL);
  transpose_cvt_kernel<<<dim3(D_MODEL / 32, (4 * D_MODEL) / 32), 256, 0, stream>>>(mlp_W1, w1T, D_MODEL, 4 * D_MODEL);
  transpose_cvt_kernel<<<dim3((4 * D_MODEL) / 32, D_MODEL / 32), 256, 0, stream>>>(mlp_W2, w2T, 4 * D_MODEL, D_MODEL);
  cvt_bf16_kernel<<<(rowsC * D_MODEL + 255) / 256, 256, 0, stream>>>(cond_tokens, ctb, rowsC * D_MODEL);

  // 1. adaLN modulation
  adaln_kernel<<<(NB * MODW + 255) / 256, 256, 0, stream>>>(cond_global, adaLN_W, adaLN_b, mod);
  // 2. LN1 + (sh_msa, sc_msa)
  ln_mod_kernel<<<rowsAll, 256, 0, stream>>>(x, norm1_w, mod, 0, D_MODEL, lnb);
  // 3. qkv projection (bf16 out)
  gemm_bf16_kernel<<<dim3(rowsAll / 128, (3 * D_MODEL) / 128), 256, 0, stream>>>(
      lnb, qkvT, qkv, rowsAll, 3 * D_MODEL, D_MODEL, D_MODEL, 3 * D_MODEL,
      nullptr, 0, 1, nullptr, nullptr, SEQ);
  // 4. RoPE
  rope_kernel<<<(NB * SEQ * NHEAD * 32) / 256, 256, 0, stream>>>(qkv, qr, kr);
  // 5. block-diff self attention
  dim3 ga(SEQ / 16, NHEAD, NB);
  attn_kernel<<<ga, 32, 0, stream>>>(qr, D_MODEL, kr, D_MODEL,
                                     qkv + 2 * D_MODEL, 3 * D_MODEL,
                                     ao, D_MODEL, SEQ, SEQ, 0, nullptr);
  // 6. attn out proj, x1 = x + g_msa * (.)
  dim3 gD(rowsAll / 128, D_MODEL / 128);
  gemm_bf16_kernel<<<gD, 256, 0, stream>>>(
      ao, aoT, x1, rowsAll, D_MODEL, D_MODEL, D_MODEL, D_MODEL,
      nullptr, 0, 0, x, mod + 2 * D_MODEL, SEQ);
  // 7. LN cross + (sh_x, sc_x)
  ln_mod_kernel<<<rowsAll, 256, 0, stream>>>(x1, normc_w, mod, 3 * D_MODEL, 4 * D_MODEL, lnb);
  // 8. qc (bf16 out)
  gemm_bf16_kernel<<<gD, 256, 0, stream>>>(
      lnb, cqT, qc, rowsAll, D_MODEL, D_MODEL, D_MODEL, D_MODEL,
      nullptr, 0, 1, nullptr, nullptr, SEQ);
  // 9/10. kc, vc from cond tokens (bf16 out)
  dim3 gC((rowsC + 127) / 128, D_MODEL / 128);
  gemm_bf16_kernel<<<gC, 256, 0, stream>>>(
      ctb, ckT, kcb, rowsC, D_MODEL, D_MODEL, D_MODEL, D_MODEL,
      nullptr, 0, 1, nullptr, nullptr, LCOND);
  gemm_bf16_kernel<<<gC, 256, 0, stream>>>(
      ctb, cvT, vcb, rowsC, D_MODEL, D_MODEL, D_MODEL, D_MODEL,
      nullptr, 0, 1, nullptr, nullptr, LCOND);
  // 11. cross attention (into ao)
  attn_kernel<<<ga, 32, 0, stream>>>(qc, D_MODEL, kcb, D_MODEL, vcb, D_MODEL,
                                     ao, D_MODEL, SEQ, LCOND, 1, cond_kv_mask);
  // 12. cross out proj, x2 = x1 + g_x * (.)
  gemm_bf16_kernel<<<gD, 256, 0, stream>>>(
      ao, coT, x2, rowsAll, D_MODEL, D_MODEL, D_MODEL, D_MODEL,
      nullptr, 0, 0, x1, mod + 5 * D_MODEL, SEQ);
  // 13. LN mlp + (sh_mlp, sc_mlp)
  ln_mod_kernel<<<rowsAll, 256, 0, stream>>>(x2, norm2_w, mod, 6 * D_MODEL, 7 * D_MODEL, lnb);
  // 14. MLP up + GELU (bf16 out)
  gemm_bf16_kernel<<<dim3(rowsAll / 128, (4 * D_MODEL) / 128), 256, 0, stream>>>(
      lnb, w1T, hb, rowsAll, 4 * D_MODEL, D_MODEL, D_MODEL, 4 * D_MODEL,
      mlp_b1, 1, 1, nullptr, nullptr, SEQ);
  // 15. MLP down, out = x2 + g_mlp * (.)  -> d_out (f32)
  gemm_bf16_kernel<<<gD, 256, 0, stream>>>(
      hb, w2T, d_out, rowsAll, D_MODEL, 4 * D_MODEL, 4 * D_MODEL, D_MODEL,
      mlp_b2, 0, 0, x2, mod + 8 * D_MODEL, SEQ);
  (void)in_sizes; (void)n_in; (void)out_size; (void)ws_size;
}
